// HCLLayer_23356032156211
// MI455X (gfx1250) — matrix operand order repaired and checked by reading, not compiled, not run on hardware
//
#include <hip/hip_runtime.h>
#include <hip/hip_bf16.h>
#include <math.h>

// Problem constants (match reference)
#define NNODES 8192
#define NEDGES 131072
#define FEATS  256
#define HIDDEN 1024
#define LN_EPS 1e-5f

typedef __attribute__((ext_vector_type(16))) __bf16 v16bf;
typedef __attribute__((ext_vector_type(8)))  float  v8f;

// Native truncation: backend selects the packed bf16 convert path.
__device__ __forceinline__ __bf16 cvbf(float x) { return (__bf16)x; }

// ---------------------------------------------------------------------------
// Degree kernels: degs = [deg_out(N) ; deg_in(N)], then rsqrt(max(d,1))
// ---------------------------------------------------------------------------
__global__ void count_degrees(const int* __restrict__ src,
                              const int* __restrict__ dst,
                              float* __restrict__ degs, int E) {
    int e = blockIdx.x * blockDim.x + threadIdx.x;
    if (e < E) {
        atomicAdd(&degs[src[e]], 1.0f);
        atomicAdd(&degs[NNODES + dst[e]], 1.0f);
    }
}

__global__ void finalize_degrees(float* __restrict__ degs, int n2) {
    int i = blockIdx.x * blockDim.x + threadIdx.x;
    if (i < n2) degs[i] = rsqrtf(fmaxf(degs[i], 1.0f));
}

// ---------------------------------------------------------------------------
// Edge scatter-add: M[dst] += H[src] * dout[src]   (one block per edge)
// ---------------------------------------------------------------------------
__global__ void edge_aggregate(const float* __restrict__ H,
                               const int* __restrict__ src,
                               const int* __restrict__ dst,
                               const float* __restrict__ dout,
                               float* __restrict__ Mbuf, int F) {
    int e = blockIdx.x;
    int s = src[e], d = dst[e];
    float sc = dout[s];
    const float* hrow = H + (size_t)s * F;
    float* mrow = Mbuf + (size_t)d * F;
    for (int t = threadIdx.x; t < F; t += blockDim.x)
        atomicAdd(&mrow[t], hrow[t] * sc);
}

// ---------------------------------------------------------------------------
// A-fragment loader: 16x32 bf16 tile (row = lane&15):
//   element e -> k = (lane/16)*16 + e  => one contiguous 16-run.
// ---------------------------------------------------------------------------
__device__ __forceinline__ v16bf load_a_frag(const float* __restrict__ Arow,
                                             int k0, int half, float rs) {
    const float4 a0 = *(const float4*)(Arow + k0 + half * 16);
    const float4 a1 = *(const float4*)(Arow + k0 + half * 16 + 4);
    const float4 a2 = *(const float4*)(Arow + k0 + half * 16 + 8);
    const float4 a3 = *(const float4*)(Arow + k0 + half * 16 + 12);
    v16bf a;
    a[0]  = cvbf(a0.x * rs); a[1]  = cvbf(a0.y * rs);
    a[2]  = cvbf(a0.z * rs); a[3]  = cvbf(a0.w * rs);
    a[4]  = cvbf(a1.x * rs); a[5]  = cvbf(a1.y * rs);
    a[6]  = cvbf(a1.z * rs); a[7]  = cvbf(a1.w * rs);
    a[8]  = cvbf(a2.x * rs); a[9]  = cvbf(a2.y * rs);
    a[10] = cvbf(a2.z * rs); a[11] = cvbf(a2.w * rs);
    a[12] = cvbf(a3.x * rs); a[13] = cvbf(a3.y * rs);
    a[14] = cvbf(a3.z * rs); a[15] = cvbf(a3.w * rs);
    return a;
}

// ---------------------------------------------------------------------------
// WMMA GEMM (NN), wide-N: one wave computes a 16 x (16*NT) strip of
// C[M,Nf] = rowscale(A[M,Kd]) @ B[Kd,Nf] + bias (opt. ReLU).
// The A fragment is loaded once per K-step and reused for NT WMMAs -> the
// A matrix is streamed from HBM only Nf/(16*NT) times (critical for the
// 256 MB attention matrix in attn@V, which cannot live in the 192 MB L2).
// B fragment: element e -> k = (lane/16)*16 + e, col = lane&15 (same k order as A).
// ---------------------------------------------------------------------------
template <int NT>
__global__ void __launch_bounds__(128)
wmma_gemm_nn(const float* __restrict__ A, const float* __restrict__ B,
             const float* __restrict__ bias, const float* __restrict__ rowscale,
             float* __restrict__ C, int M, int Kd, int Nf, int relu) {
    const int lane = threadIdx.x;            // 0..31
    const int stripsN = Nf / (16 * NT);
    const int total = (M >> 4) * stripsN;
    int strip = blockIdx.x * blockDim.y + threadIdx.y;
    if (strip >= total) return;
    const int m0 = (strip / stripsN) << 4;
    const int n0 = (strip % stripsN) * (16 * NT);
    const int half = lane >> 4;
    const int lm = lane & 15;
    const int mrow = m0 + lm;

    const float rs = rowscale ? rowscale[mrow] : 1.0f;

    v8f acc[NT];
    #pragma unroll
    for (int t = 0; t < NT; ++t) {
        const float bv = bias ? bias[n0 + t * 16 + lm] : 0.0f;
        #pragma unroll
        for (int r = 0; r < 8; ++r) acc[t][r] = bv;
    }

    const float* Arow = A + (size_t)mrow * Kd;
    for (int k0 = 0; k0 < Kd; k0 += 32) {
        if (k0 + 32 < Kd) __builtin_prefetch(Arow + k0 + 32, 0, 0);
        const v16bf a = load_a_frag(Arow, k0, half, rs);
        #pragma unroll
        for (int t = 0; t < NT; ++t) {
            const float* Bcol = B + (size_t)(k0 + (half << 4)) * Nf
                                  + n0 + t * 16 + lm;
            v16bf b;
            #pragma unroll
            for (int e = 0; e < 16; ++e)
                b[e] = cvbf(Bcol[(size_t)e * Nf]);
            acc[t] = __builtin_amdgcn_wmma_f32_16x16x32_bf16(
                false, a, false, b, (short)0, acc[t], false, false);
        }
    }

    #pragma unroll
    for (int t = 0; t < NT; ++t) {
        #pragma unroll
        for (int r = 0; r < 8; ++r) {
            float v = acc[t][r];
            if (relu) v = fmaxf(v, 0.0f);
            C[(size_t)(m0 + (half << 3) + r) * Nf + n0 + t * 16 + lm] = v;
        }
    }
}

// ---------------------------------------------------------------------------
// WMMA GEMM (NT): C[M,Nt] = scale * (A[M,Kd] @ Bt[Nt,Kd]^T)   (Q @ K^T)
// Q and K are 8 MB each -> L2-resident; one 16x16 tile per wave gives 262144
// tiles of parallelism. B fragment is contiguous in k -> float4 loads.
// ---------------------------------------------------------------------------
__global__ void __launch_bounds__(128)
wmma_gemm_nt(const float* __restrict__ A, const float* __restrict__ Bt,
             float* __restrict__ C, int M, int Nt, int Kd, float scale) {
    const int lane = threadIdx.x;
    const int tilesN = Nt >> 4;
    const int tilesTotal = (M >> 4) * tilesN;
    int tile = blockIdx.x * blockDim.y + threadIdx.y;
    if (tile >= tilesTotal) return;
    const int m0 = (tile / tilesN) << 4;
    const int n0 = (tile % tilesN) << 4;
    const int half = lane >> 4;
    const int lm = lane & 15;

    v8f acc = {};
    const float* Arow = A + (size_t)(m0 + lm) * Kd;
    const float* Brow = Bt + (size_t)(n0 + lm) * Kd;   // B[k][n] = Bt[n][k]
    for (int k0 = 0; k0 < Kd; k0 += 32) {
        const v16bf a = load_a_frag(Arow, k0, half, 1.0f);
        const float4 b0 = *(const float4*)(Brow + k0 + (half << 4));
        const float4 b1 = *(const float4*)(Brow + k0 + (half << 4) + 4);
        const float4 b2 = *(const float4*)(Brow + k0 + (half << 4) + 8);
        const float4 b3 = *(const float4*)(Brow + k0 + (half << 4) + 12);
        v16bf b;
        b[0]  = cvbf(b0.x); b[1]  = cvbf(b0.y); b[2]  = cvbf(b0.z); b[3]  = cvbf(b0.w);
        b[4]  = cvbf(b1.x); b[5]  = cvbf(b1.y); b[6]  = cvbf(b1.z); b[7]  = cvbf(b1.w);
        b[8]  = cvbf(b2.x); b[9]  = cvbf(b2.y); b[10] = cvbf(b2.z); b[11] = cvbf(b2.w);
        b[12] = cvbf(b3.x); b[13] = cvbf(b3.y); b[14] = cvbf(b3.z); b[15] = cvbf(b3.w);
        acc = __builtin_amdgcn_wmma_f32_16x16x32_bf16(
            false, a, false, b, (short)0, acc, false, false);
    }
    #pragma unroll
    for (int r = 0; r < 8; ++r)
        C[(size_t)(m0 + (half << 3) + r) * Nt + n0 + lm] = acc[r] * scale;
}

// ---------------------------------------------------------------------------
// Row softmax, in place (one block of 256 threads per row of length n)
// ---------------------------------------------------------------------------
__global__ void __launch_bounds__(256)
softmax_rows(float* __restrict__ S, int n) {
    __shared__ float red[256];
    const int tid = threadIdx.x;
    float* p = S + (size_t)blockIdx.x * n;

    float mx = -INFINITY;
    for (int j = tid; j < n; j += 256) mx = fmaxf(mx, p[j]);
    red[tid] = mx; __syncthreads();
    for (int s = 128; s > 0; s >>= 1) {
        if (tid < s) red[tid] = fmaxf(red[tid], red[tid + s]);
        __syncthreads();
    }
    mx = red[0]; __syncthreads();

    float sum = 0.0f;
    for (int j = tid; j < n; j += 256) {
        float e = __expf(p[j] - mx);
        p[j] = e;
        sum += e;
    }
    red[tid] = sum; __syncthreads();
    for (int s = 128; s > 0; s >>= 1) {
        if (tid < s) red[tid] += red[tid + s];
        __syncthreads();
    }
    float inv = 1.0f / red[0];
    for (int j = tid; j < n; j += 256) p[j] *= inv;
}

// ---------------------------------------------------------------------------
// LayerNorm over F=256 (one block of 256 threads per row), optional residual
// added AFTER normalization (reference: h = LN(attn_out) + x)
// ---------------------------------------------------------------------------
__global__ void __launch_bounds__(256)
layernorm_rows(const float* __restrict__ X, const float* __restrict__ g,
               const float* __restrict__ b, const float* __restrict__ res,
               float* __restrict__ Y) {
    __shared__ float red[256];
    const int tid = threadIdx.x;
    const size_t base = (size_t)blockIdx.x * FEATS;
    float v = X[base + tid];

    red[tid] = v; __syncthreads();
    for (int s = 128; s > 0; s >>= 1) {
        if (tid < s) red[tid] += red[tid + s];
        __syncthreads();
    }
    float mu = red[0] * (1.0f / FEATS); __syncthreads();

    float d = v - mu;
    red[tid] = d * d; __syncthreads();
    for (int s = 128; s > 0; s >>= 1) {
        if (tid < s) red[tid] += red[tid + s];
        __syncthreads();
    }
    float var = red[0] * (1.0f / FEATS);

    float y = d * rsqrtf(var + LN_EPS) * g[tid] + b[tid];
    if (res) y += res[base + tid];
    Y[base + tid] = y;
}

// ---------------------------------------------------------------------------
// Orchestration
// ---------------------------------------------------------------------------
extern "C" void kernel_launch(void* const* d_in, const int* in_sizes, int n_in,
                              void* d_out, int out_size, void* d_ws, size_t ws_size,
                              hipStream_t stream) {
    const float* x    = (const float*)d_in[0];
    const float* Wq   = (const float*)d_in[1];
    const float* bq   = (const float*)d_in[2];
    const float* Wk   = (const float*)d_in[3];
    const float* bk   = (const float*)d_in[4];
    const float* Wv   = (const float*)d_in[5];
    const float* bv   = (const float*)d_in[6];
    const float* Wo   = (const float*)d_in[7];
    const float* bo   = (const float*)d_in[8];
    const float* W1   = (const float*)d_in[9];
    const float* b1   = (const float*)d_in[10];
    const float* W2   = (const float*)d_in[11];
    const float* b2   = (const float*)d_in[12];
    const float* ln_g = (const float*)d_in[13];
    const float* ln_b = (const float*)d_in[14];
    const int*   src  = (const int*)d_in[15];
    const int*   dst  = (const int*)d_in[16];

    float* out  = (float*)d_out;                        // [N, FEATS]
    float* attn = out + (size_t)NNODES * FEATS;         // [N, N]

    // Workspace layout (floats)
    float* w    = (float*)d_ws;
    float* degs = w;                          w += 2 * NNODES;   // [dout ; din]
    float* agg  = w;                          w += (size_t)NNODES * HIDDEN;
    float* Q    = w;                          w += (size_t)NNODES * FEATS;
    float* Km   = w;                          w += (size_t)NNODES * FEATS;
    float* V    = w;                          w += (size_t)NNODES * FEATS;
    float* h    = w;                          w += (size_t)NNODES * FEATS;
    float* t1   = w;                          w += (size_t)NNODES * HIDDEN;
    float* tmp  = w;                          w += (size_t)NNODES * FEATS;
    float* dout_ = degs;
    float* din_  = degs + NNODES;

    const dim3 gblk(32, 4);   // 4 waves per block, 1 strip per wave
    auto strips = [](int M, int Nf, int nt) { return (M >> 4) * (Nf / (16 * nt)); };
    auto blocks = [](int s) { return (s + 3) / 4; };

    // 1) degrees -> rsqrt(max(deg,1))
    hipMemsetAsync(degs, 0, 2 * NNODES * sizeof(float), stream);
    count_degrees<<<(NEDGES + 255) / 256, 256, 0, stream>>>(src, dst, degs, NEDGES);
    finalize_degrees<<<(2 * NNODES + 255) / 256, 256, 0, stream>>>(degs, 2 * NNODES);

    // 2) shared aggregation of x for Q/K/V gconvs
    hipMemsetAsync(agg, 0, (size_t)NNODES * FEATS * sizeof(float), stream);
    edge_aggregate<<<NEDGES, 256, 0, stream>>>(x, src, dst, dout_, agg, FEATS);

    // 3) Q, K, V = (agg * din) @ W + b      (A is L2-resident: NT=2, max waves)
    int nbS = blocks(strips(NNODES, FEATS, 2));
    wmma_gemm_nn<2><<<nbS, gblk, 0, stream>>>(agg, Wq, bq, din_, Q, NNODES, FEATS, FEATS, 0);
    wmma_gemm_nn<2><<<nbS, gblk, 0, stream>>>(agg, Wk, bk, din_, Km, NNODES, FEATS, FEATS, 0);
    wmma_gemm_nn<2><<<nbS, gblk, 0, stream>>>(agg, Wv, bv, din_, V, NNODES, FEATS, FEATS, 0);

    // 4) scores = Q @ K^T / 16 -> attn region of d_out, softmax in place
    int nbAtt = blocks((NNODES >> 4) * (NNODES >> 4));
    wmma_gemm_nt<<<nbAtt, gblk, 0, stream>>>(Q, Km, attn, NNODES, NNODES, FEATS, 0.0625f);
    softmax_rows<<<NNODES, 256, 0, stream>>>(attn, NNODES);

    // 5) tmp = attn @ V   (A is the 256 MB attn matrix: NT=8 -> A read 2x total)
    wmma_gemm_nn<8><<<blocks(strips(NNODES, FEATS, 8)), gblk, 0, stream>>>(
        attn, V, nullptr, nullptr, tmp, NNODES, NNODES, FEATS, 0);

    // 6) attn_out = gconv(tmp, Wo, bo); h = LN(attn_out) + x
    hipMemsetAsync(agg, 0, (size_t)NNODES * FEATS * sizeof(float), stream);
    edge_aggregate<<<NEDGES, 256, 0, stream>>>(tmp, src, dst, dout_, agg, FEATS);
    wmma_gemm_nn<2><<<nbS, gblk, 0, stream>>>(agg, Wo, bo, din_, h, NNODES, FEATS, FEATS, 0);
    layernorm_rows<<<NNODES, 256, 0, stream>>>(h, ln_g, ln_b, x, h);

    // 7) t1 = relu(gconv(h, W1, b1))
    hipMemsetAsync(agg, 0, (size_t)NNODES * FEATS * sizeof(float), stream);
    edge_aggregate<<<NEDGES, 256, 0, stream>>>(h, src, dst, dout_, agg, FEATS);
    wmma_gemm_nn<8><<<blocks(strips(NNODES, HIDDEN, 8)), gblk, 0, stream>>>(
        agg, W1, b1, din_, t1, NNODES, FEATS, HIDDEN, 1);

    // 8) ff = gconv(t1, W2, b2); out = LN(ff)
    hipMemsetAsync(agg, 0, (size_t)NNODES * HIDDEN * sizeof(float), stream);
    edge_aggregate<<<NEDGES, 256, 0, stream>>>(t1, src, dst, dout_, agg, HIDDEN);
    wmma_gemm_nn<2><<<nbS, gblk, 0, stream>>>(agg, W2, b2, din_, tmp, NNODES, HIDDEN, FEATS, 0);
    layernorm_rows<<<NNODES, 256, 0, stream>>>(tmp, ln_g, ln_b, nullptr, out);
}
